// Attention_7645041787329
// MI455X (gfx1250) — matrix operand order repaired and checked by reading, not compiled, not run on hardware
//
#include <hip/hip_runtime.h>
#include <hip/hip_bf16.h>

typedef __attribute__((ext_vector_type(16))) __bf16 v16bf;
typedef __attribute__((ext_vector_type(8)))  __bf16 v8bf;
typedef __attribute__((ext_vector_type(8)))  float  v8f;
typedef int b128_t __attribute__((vector_size(16)));  // matches builtin param pointee

#define NSEQ 2048
#define DM   1024
#define HEADS 16
#define HD   64
#define SCALE_Q 0.03125f  // 1024^-0.5 exactly

#if __has_builtin(__builtin_amdgcn_global_load_async_to_lds_b128) && \
    __has_builtin(__builtin_amdgcn_s_wait_asynccnt)
#define USE_ASYNC 1
#else
#define USE_ASYNC 0
#endif

// ---- async global->LDS 16B copy (ASYNCcnt path), sync fallback ----
__device__ inline void copy16_g2l(const __bf16* g, __bf16* l) {
#if USE_ASYNC
    __builtin_amdgcn_global_load_async_to_lds_b128(
        (__attribute__((address_space(1))) b128_t*)(void*)const_cast<__bf16*>(g),
        (__attribute__((address_space(3))) b128_t*)(void*)l,
        /*offset=*/0, /*cpol=*/0);
#else
    *reinterpret_cast<v8bf*>(l) = *reinterpret_cast<const v8bf*>(g);
#endif
}
__device__ inline void async_wait0() {
#if USE_ASYNC
    __builtin_amdgcn_s_wait_asynccnt(0);
#endif
}

// ---- WMMA helper (CDNA5 16x16x32 bf16, f32 accum) ----
__device__ inline v8f wmma_bf16(v16bf a, v16bf b, v8f c) {
    return __builtin_amdgcn_wmma_f32_16x16x32_bf16(
        false, a, false, b, (short)0, c, false, false);
}

// A-fragment (16 x 32 K-slice): lane half h reads 8 bf16 at k=h*8 and k=16+h*8.
__device__ inline v16bf load_a_frag(const __bf16* rowPtr, int half) {
    v8bf lo = *reinterpret_cast<const v8bf*>(rowPtr + half * 8);
    v8bf hi = *reinterpret_cast<const v8bf*>(rowPtr + 16 + half * 8);
    v16bf f;
#pragma unroll
    for (int i = 0; i < 8; ++i) { f[i] = lo[i]; f[i + 8] = hi[i]; }
    return f;
}
// B-fragment (32 K x 16 N slice): lane half h reads 8 bf16 at k=h*8 and k=16+h*8.
__device__ inline v16bf load_b_frag(const __bf16* rowPtr, int half) {
    v8bf lo = *reinterpret_cast<const v8bf*>(rowPtr + half * 8);
    v8bf hi = *reinterpret_cast<const v8bf*>(rowPtr + 16 + half * 8);
    v16bf f;
#pragma unroll
    for (int i = 0; i < 8; ++i) { f[i] = lo[i]; f[i + 8] = hi[i]; }
    return f;
}

// ---- fp32 -> bf16 convert ----
__global__ void cvt_f32_bf16(const float* __restrict__ in, __bf16* __restrict__ out, long n) {
    long i = (long)blockIdx.x * blockDim.x + threadIdx.x;
    long stride = (long)gridDim.x * blockDim.x;
    for (; i < n; i += stride) out[i] = (__bf16)in[i];
}

// ---- Double-buffered bf16 GEMM: C[M,N] = A[M,K] @ W[N,K]^T ----
// 256 thr (8 waves); workgroup tile 128(M) x 64(N); K-step 64; async staging.
// mode 0: QKV epilogue (scatter q/k/v [b,h,n,d] bf16, Q scaled 1/32)
// mode 1: out epilogue (add bias, fp32)
__global__ __launch_bounds__(256)
void gemm_bf16(const __bf16* __restrict__ A, const __bf16* __restrict__ W,
               int Kdim, int mode,
               __bf16* __restrict__ qout, __bf16* __restrict__ kout,
               __bf16* __restrict__ vout,
               const float* __restrict__ bias, float* __restrict__ out, int Nout) {
    __shared__ __bf16 Alds[2][128 * 72];  // 64-wide K tile, pad to 72 (144B rows)
    __shared__ __bf16 Wlds[2][64 * 72];

    const int tid  = threadIdx.x;
    const int wave = tid >> 5;
    const int lane = tid & 31;
    const int lg   = lane & 15;
    const int half = lane >> 4;
    const int nBase = blockIdx.x * 64;
    const int mBase = blockIdx.y * 128;
    const int mw = (wave >> 1) * 32;
    const int nw = (wave & 1) * 32;

    v8f acc[2][2];
#pragma unroll
    for (int i = 0; i < 2; ++i)
#pragma unroll
        for (int j = 0; j < 2; ++j)
#pragma unroll
            for (int r = 0; r < 8; ++r) acc[i][j][r] = 0.0f;

    // stage tile (128x64 A chunks + 64x64 W chunks, 16B each)
    auto stage = [&](int buf, int k0) {
#pragma unroll
        for (int idx = tid, rep = 0; rep < 4; ++rep, idx += 256) {  // 1024 A-chunks
            int row = idx >> 3, c = (idx & 7) * 8;
            copy16_g2l(A + (long)(mBase + row) * Kdim + k0 + c, &Alds[buf][row * 72 + c]);
        }
#pragma unroll
        for (int idx = tid, rep = 0; rep < 2; ++rep, idx += 256) {  // 512 W-chunks
            int row = idx >> 3, c = (idx & 7) * 8;
            copy16_g2l(W + (long)(nBase + row) * Kdim + k0 + c, &Wlds[buf][row * 72 + c]);
        }
    };

    stage(0, 0);
    async_wait0();
    __syncthreads();

    int cur = 0;
    for (int k0 = 0; k0 < Kdim; k0 += 64) {
        if (k0 + 64 < Kdim) stage(cur ^ 1, k0 + 64);  // async prefetch next tile

#pragma unroll
        for (int kk = 0; kk < 2; ++kk) {              // two 32-deep K slices
            v16bf af[2], bfr[2];
#pragma unroll
            for (int i = 0; i < 2; ++i)
                af[i] = load_a_frag(&Alds[cur][(mw + i * 16 + lg) * 72 + kk * 32], half);
#pragma unroll
            for (int j = 0; j < 2; ++j)
                bfr[j] = load_b_frag(&Wlds[cur][(nw + j * 16 + lg) * 72 + kk * 32], half);
#pragma unroll
            for (int i = 0; i < 2; ++i)
#pragma unroll
                for (int j = 0; j < 2; ++j)
                    acc[i][j] = wmma_bf16(af[i], bfr[j], acc[i][j]);
        }
        async_wait0();
        __syncthreads();
        cur ^= 1;
    }

    // epilogue; C layout: row = r + 8*half, col = lane%16 within each 16x16 tile
#pragma unroll
    for (int i = 0; i < 2; ++i)
#pragma unroll
        for (int j = 0; j < 2; ++j)
#pragma unroll
            for (int r = 0; r < 8; ++r) {
                int m   = mBase + mw + i * 16 + r + 8 * half;
                int col = nBase + nw + j * 16 + lg;
                float val = acc[i][j][r];
                if (mode == 0) {
                    int c = col >> 10;
                    int within = col & 1023;
                    int hh = within >> 6;
                    int d  = within & 63;
                    int bb = m >> 11;
                    int n  = m & 2047;
                    long dst = (((long)(bb * HEADS + hh)) * NSEQ + n) * HD + d;
                    if (c == 0)      qout[dst] = (__bf16)(val * SCALE_Q);
                    else if (c == 1) kout[dst] = (__bf16)val;
                    else             vout[dst] = (__bf16)val;
                } else {
                    out[(long)m * Nout + col] = val + bias[col];
                }
            }
}

// ---- Flash attention: block per (64-row qtile, h, b); 128 thr = 4 waves ----
__global__ __launch_bounds__(128)
void flash_attn(const __bf16* __restrict__ q, const __bf16* __restrict__ k,
                const __bf16* __restrict__ v, __bf16* __restrict__ ctx) {
    __shared__ __bf16 Klds[64 * 72];      // [j][d]
    __shared__ __bf16 Vt[64 * 72];        // [d][j] transposed
    __shared__ __bf16 Plds[4 * 16 * 72];  // per-wave P staging

    const int tid  = threadIdx.x;
    const int wave = tid >> 5;
    const int lane = tid & 31;
    const int lg   = lane & 15;
    const int half = lane >> 4;
    const int qtile = blockIdx.x;
    const int h = blockIdx.y;
    const int b = blockIdx.z;
    const long bh = (long)(b * HEADS + h);
    const __bf16* kbh = k + bh * NSEQ * HD;
    const __bf16* vbh = v + bh * NSEQ * HD;
    const int qbase = qtile * 64 + wave * 16;

    const __bf16* qrow = q + (bh * NSEQ + qbase + lg) * HD;
    v16bf qfrag[2];
    qfrag[0] = load_a_frag(qrow, half);
    qfrag[1] = load_a_frag(qrow + 32, half);

    v8f oacc[4];
    float mrow[8], lrow[8];
#pragma unroll
    for (int t = 0; t < 4; ++t)
#pragma unroll
        for (int r = 0; r < 8; ++r) oacc[t][r] = 0.0f;
#pragma unroll
    for (int r = 0; r < 8; ++r) { mrow[r] = -1e30f; lrow[r] = 0.0f; }

    __bf16* pw = Plds + wave * 16 * 72;

    for (int it = 0; it < NSEQ / 64; ++it) {
        // K tile: async copy, row-major (512 x 16B chunks over 4 reps)
        const __bf16* kg = kbh + (long)it * 64 * HD;
#pragma unroll
        for (int idx = tid, rep = 0; rep < 4; ++rep, idx += 128) {
            int row = idx >> 3, c = (idx & 7) * 8;
            copy16_g2l(kg + row * 64 + c, Klds + row * 72 + c);
        }
        // V tile: manual transpose into LDS
        const __bf16* vg = vbh + (long)it * 64 * HD;
        for (int idx = tid; idx < 512; idx += 128) {
            int row = idx >> 3, c = (idx & 7) * 8;
            v8bf vv = *reinterpret_cast<const v8bf*>(vg + row * 64 + c);
#pragma unroll
            for (int i = 0; i < 8; ++i) Vt[(c + i) * 72 + row] = vv[i];
        }
        async_wait0();
        __syncthreads();

        // S = Q K^T  (Q pre-scaled by 1/32)
        v8f sacc[4];
#pragma unroll
        for (int t = 0; t < 4; ++t) {
#pragma unroll
            for (int r = 0; r < 8; ++r) sacc[t][r] = 0.0f;
#pragma unroll
            for (int kk = 0; kk < 2; ++kk) {
                v16bf bf = load_b_frag(Klds + (t * 16 + lg) * 72 + kk * 32, half);
                sacc[t] = wmma_bf16(qfrag[kk], bf, sacc[t]);
            }
        }

        // online softmax (row = r + 8*half; 16-lane group reductions)
#pragma unroll
        for (int r = 0; r < 8; ++r) {
            float rm = fmaxf(fmaxf(sacc[0][r], sacc[1][r]), fmaxf(sacc[2][r], sacc[3][r]));
            rm = fmaxf(rm, __shfl_xor(rm, 1));
            rm = fmaxf(rm, __shfl_xor(rm, 2));
            rm = fmaxf(rm, __shfl_xor(rm, 4));
            rm = fmaxf(rm, __shfl_xor(rm, 8));
            float mnew = fmaxf(mrow[r], rm);
            float corr = __expf(mrow[r] - mnew);
            mrow[r] = mnew;
            float rs = 0.0f;
#pragma unroll
            for (int t = 0; t < 4; ++t) {
                float p = __expf(sacc[t][r] - mnew);
                sacc[t][r] = p;
                rs += p;
            }
            rs += __shfl_xor(rs, 1);
            rs += __shfl_xor(rs, 2);
            rs += __shfl_xor(rs, 4);
            rs += __shfl_xor(rs, 8);
            lrow[r] = lrow[r] * corr + rs;
#pragma unroll
            for (int t = 0; t < 4; ++t) oacc[t][r] *= corr;
#pragma unroll
            for (int t = 0; t < 4; ++t)
                pw[(r + 8 * half) * 72 + t * 16 + lg] = (__bf16)sacc[t][r];
        }

        // O += P (16x64) x V (64x64)
#pragma unroll
        for (int t = 0; t < 4; ++t)
#pragma unroll
            for (int kk = 0; kk < 2; ++kk) {
                v16bf af = load_a_frag(pw + lg * 72 + kk * 32, half);
                v16bf bf = load_b_frag(Vt + (t * 16 + lg) * 72 + kk * 32, half);
                oacc[t] = wmma_bf16(af, bf, oacc[t]);
            }
        __syncthreads();
    }

#pragma unroll
    for (int r = 0; r < 8; ++r) {
        float inv = 1.0f / lrow[r];
        int row = qbase + r + 8 * half;
        long base = ((long)(b * NSEQ + row)) * DM + h * HD;
#pragma unroll
        for (int t = 0; t < 4; ++t)
            ctx[base + t * 16 + lg] = (__bf16)(oacc[t][r] * inv);
    }
}

extern "C" void kernel_launch(void* const* d_in, const int* in_sizes, int n_in,
                              void* d_out, int out_size, void* d_ws, size_t ws_size,
                              hipStream_t stream) {
    const float* x     = (const float*)d_in[0];   // [4,2048,1024]
    const float* w_qkv = (const float*)d_in[1];   // [3072,1024]
    const float* w_out = (const float*)d_in[2];   // [1024,1024]
    const float* b_out = (const float*)d_in[3];   // [1024]
    float* out = (float*)d_out;

    char* ws = (char*)d_ws;
    __bf16* xb  = (__bf16*)(ws);                       // 16 MB
    __bf16* wqb = (__bf16*)(ws + 16777216L);           // 6 MB
    __bf16* wob = (__bf16*)(ws + 23068672L);           // 2 MB
    __bf16* qb  = (__bf16*)(ws + 25165824L);           // 16 MB
    __bf16* kb  = (__bf16*)(ws + 41943040L);           // 16 MB
    __bf16* vb  = (__bf16*)(ws + 58720256L);           // 16 MB
    __bf16* cb  = (__bf16*)(ws + 75497472L);           // 16 MB

    cvt_f32_bf16<<<2048, 256, 0, stream>>>(x, xb, 8388608L);
    cvt_f32_bf16<<<1024, 256, 0, stream>>>(w_qkv, wqb, 3145728L);
    cvt_f32_bf16<<<512, 256, 0, stream>>>(w_out, wob, 1048576L);

    gemm_bf16<<<dim3(3072 / 64, 8192 / 128), 256, 0, stream>>>(
        xb, wqb, 1024, 0, qb, kb, vb, nullptr, nullptr, 3072);

    flash_attn<<<dim3(NSEQ / 64, HEADS, 4), 128, 0, stream>>>(qb, kb, vb, cb);

    gemm_bf16<<<dim3(1024 / 64, 8192 / 128), 256, 0, stream>>>(
        cb, wob, 1024, 1, nullptr, nullptr, nullptr, b_out, out, 1024);
}
